// Decoder_8117488190024
// MI455X (gfx1250) — compile-verified, matrix operand order checked by reading, not run on hardware
//
#include <hip/hip_runtime.h>
#include <hip/hip_bf16.h>

typedef __bf16 bf16;
typedef __attribute__((ext_vector_type(16))) __bf16 v16bf;
typedef __attribute__((ext_vector_type(8)))  __bf16 v8bf;
typedef __attribute__((ext_vector_type(8)))  float  v8f;

#define CDIM   768
#define HEADS  12
#define DHEAD  64
#define TSEQ   1024
#define RLORA  8

__device__ inline v8f wmma_bf16f32(v16bf a, v16bf b, v8f c) {
  // D = A(16x32 bf16) * B(32x16 bf16) + C(16x16 f32)
  return __builtin_amdgcn_wmma_f32_16x16x32_bf16(false, a, false, b, (short)0, c,
                                                 false, false);
}

__device__ inline float gelu_tanh(float x) {
  float x3 = x * x * x;
  return 0.5f * x * (1.0f + tanhf(0.7978845608028654f * (x + 0.044715f * x3)));
}

__device__ inline float wave_red_sum(float v) {
#pragma unroll
  for (int m = 16; m >= 1; m >>= 1) v += __shfl_xor(v, m, 32);
  return v;
}
__device__ inline float half_red_max(float v) {
#pragma unroll
  for (int m = 8; m >= 1; m >>= 1) v = fmaxf(v, __shfl_xor(v, m, 16));
  return v;
}
__device__ inline float half_red_sum(float v) {
#pragma unroll
  for (int m = 8; m >= 1; m >>= 1) v += __shfl_xor(v, m, 16);
  return v;
}

// ---------------------------------------------------------------- embedding
__global__ void k_embed(const int* __restrict__ x, const float* __restrict__ img,
                        const float* __restrict__ wte, const float* __restrict__ wpe,
                        float* __restrict__ h) {
  int idx = blockIdx.x * blockDim.x + threadIdx.x;
  if (idx >= 2 * TSEQ * CDIM) return;
  int c = idx % CDIM;
  int t = (idx / CDIM) % TSEQ;
  int b = idx / (CDIM * TSEQ);
  float v;
  if (t < 64) v = img[((size_t)b * 64 + t) * CDIM + c];
  else        v = wte[(size_t)x[b * 960 + (t - 64)] * CDIM + c];
  h[idx] = v + wpe[(size_t)t * CDIM + c];
}

// ---------------------------------------------------------------- layernorm
__global__ __launch_bounds__(128) void k_layernorm(
    const float* __restrict__ X, const float* __restrict__ g,
    const float* __restrict__ bta, float* __restrict__ outF,
    bf16* __restrict__ outB, int Csz) {
  int row  = blockIdx.x * 4 + (threadIdx.x >> 5);
  int lane = threadIdx.x & 31;
  const float* xr = X + (size_t)row * Csz;
  float s = 0.f, s2 = 0.f;
  for (int k = lane; k < Csz; k += 32) { float v = xr[k]; s += v; s2 += v * v; }
  s  = wave_red_sum(s);
  s2 = wave_red_sum(s2);
  float mean = s / (float)Csz;
  float var  = s2 / (float)Csz - mean * mean;
  float inv  = rsqrtf(var + 1e-5f);
  for (int k = lane; k < Csz; k += 32) {
    float v = (xr[k] - mean) * inv * g[k] + bta[k];
    size_t oi = (size_t)row * Csz + k;
    if (outF) outF[oi] = v;
    if (outB) outB[oi] = (bf16)v;
  }
}

// ------------------------------------------------------------- LoRA down (rank 8)
__global__ __launch_bounds__(128) void k_lora_down(
    const float* __restrict__ Z, const float* __restrict__ Am,
    float* __restrict__ tmp, int Ksz) {
  int row  = blockIdx.x * 4 + (threadIdx.x >> 5);
  int lane = threadIdx.x & 31;
  const float* z = Z + (size_t)row * Ksz;
  float acc[RLORA];
#pragma unroll
  for (int r = 0; r < RLORA; r++) acc[r] = 0.f;
  for (int k = lane; k < Ksz; k += 32) {
    float zv = z[k];
#pragma unroll
    for (int r = 0; r < RLORA; r++) acc[r] += zv * Am[(size_t)r * Ksz + k];
  }
#pragma unroll
  for (int r = 0; r < RLORA; r++) acc[r] = wave_red_sum(acc[r]);
  if (lane == 0) {
#pragma unroll
    for (int r = 0; r < RLORA; r++) tmp[(size_t)row * RLORA + r] = acc[r];
  }
}

// ---------------------------------------------------------------- WMMA GEMM
// out[M,N] = A[M,K](bf16) @ W[N,K]^T(f32->bf16) + bias + loraT@loraB^T
// Block tile 64x128, 4 waves; each wave: 64x32 outputs = acc[4][2],
// 8 v_wmma per K-step. A-tile staged memory->LDS with async-to-LDS ops.
__global__ __launch_bounds__(128) void k_gemm(
    const bf16* __restrict__ A, const float* __restrict__ W,
    const float* __restrict__ bias, const float* __restrict__ loraT,
    const float* __restrict__ loraB, const float* __restrict__ residual,
    float* __restrict__ outF, bf16* __restrict__ outB,
    int Msz, int Nsz, int Ksz, int act) {
  __shared__ bf16 As[64 * 32];    // 4 KB
  __shared__ bf16 Ws[128 * 32];   // 8 KB
  const int tid   = threadIdx.x;
  const int lane  = tid & 31;
  const int wave  = tid >> 5;
  const int mbase = blockIdx.y * 64;
  const int nbase = blockIdx.x * 128;
  const int wn    = wave * 32;
  const int half  = lane >> 4;
  const int c16   = lane & 15;

  v8f z8 = {0.f, 0.f, 0.f, 0.f, 0.f, 0.f, 0.f, 0.f};
  v8f acc[4][2];
#pragma unroll
  for (int ti = 0; ti < 4; ti++) { acc[ti][0] = z8; acc[ti][1] = z8; }

  const int aRow = tid >> 1;        // 0..63
  const int aCol = (tid & 1) * 16;  // 0 or 16

  for (int k0 = 0; k0 < Ksz; k0 += 32) {
    { // stage A tile: async copy global(bf16) -> LDS, no VGPR round-trip.
      // INST_OFFSET applies to both LDS and global sides (ISA 15.18.3),
      // so one address pair covers both 16B halves of this thread's 32B.
      const bf16* g = A + (size_t)(mbase + aRow) * Ksz + k0 + aCol;
      unsigned lds = (unsigned)(uintptr_t)(&As[aRow * 32 + aCol]);
      asm volatile(
          "global_load_async_to_lds_b128 %0, %1, off\n\t"
          "global_load_async_to_lds_b128 %0, %1, off offset:16"
          :: "v"(lds), "v"(g) : "memory");
    }
    // stage W tile with f32->bf16 conversion; 2 rows/thread; guard N edge
#pragma unroll
    for (int rr = 0; rr < 2; rr++) {
      int row = aRow + rr * 64;
      int n = nbase + row;
      bf16* dst = &Ws[row * 32 + aCol];
      if (n < Nsz) {
        const float* g = W + (size_t)n * Ksz + k0 + aCol;
#pragma unroll
        for (int i = 0; i < 16; i++) dst[i] = (bf16)g[i];
      } else {
#pragma unroll
        for (int i = 0; i < 16; i++) dst[i] = (bf16)0.f;
      }
    }
    if (k0 + 32 < Ksz)
      __builtin_prefetch(A + (size_t)(mbase + aRow) * Ksz + k0 + 32, 0, 1);
    asm volatile("s_wait_asynccnt 0x0" ::: "memory");
    __syncthreads();

    // fragments per ISA 16-bit A layout: lane row = l%16,
    // K chunk = (l/16)*8 in elems 0..7, +16 in elems 8..15
    v16bf af[4], wf[2];
#pragma unroll
    for (int ti = 0; ti < 4; ti++) {
      const bf16* pa = &As[(ti * 16 + c16) * 32 + half * 8];
      v8bf lo = *(const v8bf*)pa;
      v8bf hi = *(const v8bf*)(pa + 16);
      af[ti] = __builtin_shufflevector(lo, hi, 0, 1, 2, 3, 4, 5, 6, 7,
                                       8, 9, 10, 11, 12, 13, 14, 15);
    }
#pragma unroll
    for (int tj = 0; tj < 2; tj++) {
      const bf16* pw = &Ws[(wn + tj * 16 + c16) * 32 + half * 8];
      v8bf lo = *(const v8bf*)pw;
      v8bf hi = *(const v8bf*)(pw + 16);
      wf[tj] = __builtin_shufflevector(lo, hi, 0, 1, 2, 3, 4, 5, 6, 7,
                                       8, 9, 10, 11, 12, 13, 14, 15);
    }
#pragma unroll
    for (int ti = 0; ti < 4; ti++)
#pragma unroll
      for (int tj = 0; tj < 2; tj++)
        acc[ti][tj] = wmma_bf16f32(af[ti], wf[tj], acc[ti][tj]);
    __syncthreads();
  }

  // epilogue: C/D layout lane l, vgpr r -> (M = r + 8*(l/16), N = l%16)
#pragma unroll
  for (int ti = 0; ti < 4; ti++) {
#pragma unroll
    for (int tj = 0; tj < 2; tj++) {
      int n = nbase + wn + tj * 16 + c16;
      if (n >= Nsz) continue;
      float bia = bias ? bias[n] : 0.f;
      float lb[RLORA];
      if (loraB) {
#pragma unroll
        for (int j = 0; j < RLORA; j++) lb[j] = loraB[(size_t)n * RLORA + j];
      }
#pragma unroll
      for (int r = 0; r < 8; r++) {
        int m = mbase + ti * 16 + r + half * 8;
        float v = acc[ti][tj][r] + bia;
        if (loraB) {
          const float* t = loraT + (size_t)m * RLORA;
#pragma unroll
          for (int j = 0; j < RLORA; j++) v += t[j] * lb[j];
        }
        if (act) v = gelu_tanh(v);
        if (residual) v += residual[(size_t)m * Nsz + n];
        size_t oi = (size_t)m * Nsz + n;
        if (outF) outF[oi] = v;
        if (outB) outB[oi] = (bf16)v;
      }
    }
  }
}

// --------------------------------------------------- flash-style WMMA attention
// qkv: [B, T, 3C] f32. One wave owns 16 query rows; streams 32-key tiles.
__global__ __launch_bounds__(128) void k_attn(const float* __restrict__ qkv,
                                              float* __restrict__ oF,
                                              bf16* __restrict__ oB) {
  __shared__ bf16 Pl[4][16 * 32];
  const int wave = threadIdx.x >> 5;
  const int lane = threadIdx.x & 31;
  const int half = lane >> 4;
  const int c16  = lane & 15;
  const int b    = blockIdx.y / HEADS;
  const int h    = blockIdx.y % HEADS;
  const int q0   = blockIdx.x * 64 + wave * 16;

  const float* base = qkv + (size_t)b * TSEQ * (3 * CDIM);
  const float* Qp = base + h * DHEAD;
  const float* Kp = base + CDIM + h * DHEAD;
  const float* Vp = base + 2 * CDIM + h * DHEAD;

  // Q A-fragments for d-chunks [0,32) and [32,64); 1/sqrt(64) folded in
  v16bf aq[2];
  {
    const float* qr = Qp + (size_t)(q0 + c16) * (3 * CDIM);
#pragma unroll
    for (int ch = 0; ch < 2; ch++) {
      int kb = ch * 32 + half * 8;
      v16bf f;
#pragma unroll
      for (int i = 0; i < 8; i++) f[i]     = (bf16)(qr[kb + i] * 0.125f);
#pragma unroll
      for (int i = 0; i < 8; i++) f[8 + i] = (bf16)(qr[kb + 16 + i] * 0.125f);
      aq[ch] = f;
    }
  }

  v8f z8 = {0.f, 0.f, 0.f, 0.f, 0.f, 0.f, 0.f, 0.f};
  v8f oacc[4] = {z8, z8, z8, z8};
  float rmax[8], rsum[8];
#pragma unroll
  for (int r = 0; r < 8; r++) { rmax[r] = -3.0e38f; rsum[r] = 0.f; }

  const int ktiles = blockIdx.x * 2 + 2;  // uniform across block (causal max)
  for (int kt = 0; kt < ktiles; kt++) {
    const int kb0 = kt * 32;
    v8f S[2] = {z8, z8};
#pragma unroll
    for (int t2 = 0; t2 < 2; t2++) {
      const float* kr = Kp + (size_t)(kb0 + t2 * 16 + c16) * (3 * CDIM);
#pragma unroll
      for (int ch = 0; ch < 2; ch++) {
        int kb = ch * 32 + half * 8;
        v16bf f;
#pragma unroll
        for (int i = 0; i < 8; i++) f[i]     = (bf16)kr[kb + i];
#pragma unroll
        for (int i = 0; i < 8; i++) f[8 + i] = (bf16)kr[kb + 16 + i];
        S[t2] = wmma_bf16f32(aq[ch], f, S[t2]);
      }
    }
    // causal mask in C/D layout
#pragma unroll
    for (int t2 = 0; t2 < 2; t2++) {
      int ki = kb0 + t2 * 16 + c16;
#pragma unroll
      for (int r = 0; r < 8; r++) {
        int qi = q0 + r + half * 8;
        if (ki > qi) S[t2][r] = -3.0e38f;
      }
    }
    // online softmax: rows live in 16-lane halves
#pragma unroll
    for (int r = 0; r < 8; r++) {
      float mx = fmaxf(S[0][r], S[1][r]);
      mx = half_red_max(mx);
      float nm = fmaxf(rmax[r], mx);
      float corr = __expf(rmax[r] - nm);
      float p0 = __expf(S[0][r] - nm); if (S[0][r] < -1.0e37f) p0 = 0.f;
      float p1 = __expf(S[1][r] - nm); if (S[1][r] < -1.0e37f) p1 = 0.f;
      S[0][r] = p0; S[1][r] = p1;
      float ls = half_red_sum(p0 + p1);
      rsum[r] = rsum[r] * corr + ls;
      rmax[r] = nm;
#pragma unroll
      for (int j = 0; j < 4; j++) oacc[j][r] *= corr;
    }
    __syncthreads();
    // P: C/D layout -> LDS row-major -> A-fragment layout
    bf16* pb = &Pl[wave][0];
#pragma unroll
    for (int t2 = 0; t2 < 2; t2++)
#pragma unroll
      for (int r = 0; r < 8; r++)
        pb[(r + half * 8) * 32 + t2 * 16 + c16] = (bf16)S[t2][r];
    __syncthreads();
    v16bf ap;
    {
      const bf16* pr = pb + c16 * 32 + half * 8;
      v8bf lo = *(const v8bf*)pr;
      v8bf hi = *(const v8bf*)(pr + 16);
      ap = __builtin_shufflevector(lo, hi, 0, 1, 2, 3, 4, 5, 6, 7,
                                   8, 9, 10, 11, 12, 13, 14, 15);
    }
    // O += P(16x32) @ V(32x64), 4 d-chunks of 16
#pragma unroll
    for (int j = 0; j < 4; j++) {
      const float* vc = Vp + j * 16 + c16;
      int kb2 = half * 8;
      v16bf f;
#pragma unroll
      for (int i = 0; i < 8; i++)
        f[i] = (bf16)vc[(size_t)(kb0 + kb2 + i) * (3 * CDIM)];
#pragma unroll
      for (int i = 0; i < 8; i++)
        f[8 + i] = (bf16)vc[(size_t)(kb0 + kb2 + 16 + i) * (3 * CDIM)];
      oacc[j] = wmma_bf16f32(ap, f, oacc[j]);
    }
  }

#pragma unroll
  for (int j = 0; j < 4; j++) {
#pragma unroll
    for (int r = 0; r < 8; r++) {
      int qi = q0 + r + half * 8;
      float val = oacc[j][r] / rsum[r];
      size_t oi = ((size_t)b * TSEQ + qi) * CDIM + h * DHEAD + j * 16 + c16;
      oF[oi] = val;
      oB[oi] = (bf16)val;
    }
  }
}

// ------------------------------------------------------------------ launcher
extern "C" void kernel_launch(void* const* d_in, const int* in_sizes, int n_in,
                              void* d_out, int out_size, void* d_ws, size_t ws_size,
                              hipStream_t stream) {
  (void)in_sizes; (void)n_in; (void)out_size; (void)ws_size;
  const int*   x      = (const int*)d_in[0];
  const float* img    = (const float*)d_in[1];
  const float* wte    = (const float*)d_in[2];
  const float* wpe    = (const float*)d_in[3];
  const float* lnf_g  = (const float*)d_in[4];
  const float* lnf_b  = (const float*)d_in[5];
  const float* ln1_g  = (const float*)d_in[6];
  const float* ln1_b  = (const float*)d_in[7];
  const float* ln2_g  = (const float*)d_in[8];
  const float* ln2_b  = (const float*)d_in[9];
  const float* attn_w = (const float*)d_in[10];
  const float* attn_b = (const float*)d_in[11];
  const float* attn_A = (const float*)d_in[12];
  const float* attn_B = (const float*)d_in[13];
  const float* proj_w = (const float*)d_in[14];
  const float* proj_b = (const float*)d_in[15];
  const float* proj_A = (const float*)d_in[16];
  const float* proj_B = (const float*)d_in[17];
  const float* fc_w   = (const float*)d_in[18];
  const float* fc_b   = (const float*)d_in[19];
  const float* fc_A   = (const float*)d_in[20];
  const float* fc_B   = (const float*)d_in[21];
  const float* mp_w   = (const float*)d_in[22];
  const float* mp_b   = (const float*)d_in[23];
  const float* mp_A   = (const float*)d_in[24];
  const float* mp_B   = (const float*)d_in[25];
  float* out = (float*)d_out;

  const int C = 768, M = 2048, R = 8, V = 50257;
  char* ws = (char*)d_ws;
  size_t off = 0;
  auto carve = [&](size_t bytes) {
    char* p = ws + off;
    off = (off + bytes + 255) & ~(size_t)255;
    return p;
  };
  float* hbuf  = (float*)carve((size_t)M * C * 4);
  float* lnF   = (float*)carve((size_t)M * C * 4);
  bf16*  lnB   = (bf16*) carve((size_t)M * C * 2);
  float* qkvF  = (float*)carve((size_t)M * 3 * C * 4);
  float* attF  = (float*)carve((size_t)M * C * 4);
  bf16*  attB  = (bf16*) carve((size_t)M * C * 2);
  float* mlpF  = (float*)carve((size_t)M * 4 * C * 4);
  bf16*  mlpB  = (bf16*) carve((size_t)M * 4 * C * 2);
  float* loraT = (float*)carve((size_t)M * R * 4);

  dim3 blk(128);
  k_embed<<<dim3((2 * TSEQ * CDIM + 255) / 256), dim3(256), 0, stream>>>(
      x, img, wte, wpe, hbuf);

  for (int l = 0; l < 6; l++) {
    k_layernorm<<<dim3(M / 4), blk, 0, stream>>>(hbuf, ln1_g + l * C, ln1_b + l * C,
                                                 lnF, lnB, C);
    k_lora_down<<<dim3(M / 4), blk, 0, stream>>>(lnF, attn_A + (size_t)l * R * C,
                                                 loraT, C);
    k_gemm<<<dim3((3 * C) / 128, M / 64), blk, 0, stream>>>(
        lnB, attn_w + (size_t)l * 3 * C * C, attn_b + (size_t)l * 3 * C,
        loraT, attn_B + (size_t)l * 3 * C * R, nullptr, qkvF, nullptr,
        M, 3 * C, C, 0);
    k_attn<<<dim3(TSEQ / 64, 2 * HEADS), blk, 0, stream>>>(qkvF, attF, attB);
    k_lora_down<<<dim3(M / 4), blk, 0, stream>>>(attF, proj_A + (size_t)l * R * C,
                                                 loraT, C);
    k_gemm<<<dim3(C / 128, M / 64), blk, 0, stream>>>(
        attB, proj_w + (size_t)l * C * C, proj_b + (size_t)l * C,
        loraT, proj_B + (size_t)l * C * R, hbuf, hbuf, nullptr, M, C, C, 0);
    k_layernorm<<<dim3(M / 4), blk, 0, stream>>>(hbuf, ln2_g + l * C, ln2_b + l * C,
                                                 lnF, lnB, C);
    k_lora_down<<<dim3(M / 4), blk, 0, stream>>>(lnF, fc_A + (size_t)l * R * C,
                                                 loraT, C);
    k_gemm<<<dim3((4 * C) / 128, M / 64), blk, 0, stream>>>(
        lnB, fc_w + (size_t)l * 4 * C * C, fc_b + (size_t)l * 4 * C,
        loraT, fc_B + (size_t)l * 4 * C * R, nullptr, mlpF, mlpB,
        M, 4 * C, C, 1);
    k_lora_down<<<dim3(M / 4), blk, 0, stream>>>(mlpF, mp_A + (size_t)l * R * 4 * C,
                                                 loraT, 4 * C);
    k_gemm<<<dim3(C / 128, M / 64), blk, 0, stream>>>(
        mlpB, mp_w + (size_t)l * C * 4 * C, mp_b + (size_t)l * C,
        loraT, mp_B + (size_t)l * C * R, hbuf, hbuf, nullptr, M, C, 4 * C, 0);
  }

  k_layernorm<<<dim3(M / 4), blk, 0, stream>>>(hbuf, lnf_g, lnf_b, nullptr, lnB, C);
  k_gemm<<<dim3((V + 127) / 128, M / 64), blk, 0, stream>>>(
      lnB, wte, nullptr, nullptr, nullptr, nullptr, out, nullptr, M, V, C, 0);
}
